// MultiheadLatentAttention_5368709120666
// MI455X (gfx1250) — compile-verified, matrix operand order checked by reading, not run on hardware
//
#include <hip/hip_runtime.h>
#include <hip/hip_bf16.h>

// ---------------------------------------------------------------------------
// Sliding-window GQA attention for MI455X (gfx1250, wave32, WMMA).
// fp32->bf16 convert, fused QKV GEMM (bf16 WMMA, fp32 accum, async LDS copy,
// double-buffered), fused flash-style windowed attention (WMMA QK^T + WMMA PV,
// online softmax), output-projection GEMM -> fp32.
// ---------------------------------------------------------------------------

typedef unsigned short u16;
typedef __attribute__((ext_vector_type(16))) __bf16 v16bf;
typedef __attribute__((ext_vector_type(8)))  float  v8f;

#define HQ   16
#define HKV  4
#define HD   128
#define WIN  512
#define SEQ  8192
#define DIM  2048
#define NQK  (HQ * HD + 2 * HKV * HD)   /* 3072: q|k|v packed per token */

// ---- CDNA5 async global->LDS copy (ASYNCcnt) ------------------------------
#if defined(__gfx1250__) && __has_builtin(__builtin_amdgcn_global_load_async_to_lds_b128)
#define ASYNC_COPY 1
__device__ __forceinline__ void async_cp16(const u16* g, u16* l) {
  typedef int vi4 __attribute__((vector_size(16)));
  typedef __attribute__((address_space(1))) const u16* gu16p;
  typedef __attribute__((address_space(3))) u16*       lu16p;
  typedef __attribute__((address_space(1))) vi4*       g4p;
  typedef __attribute__((address_space(3))) vi4*       l4p;
  gu16p g1 = (gu16p)g;   // addrspacecast generic -> global
  lu16p l3 = (lu16p)l;   // addrspacecast generic -> LDS
  __builtin_amdgcn_global_load_async_to_lds_b128((g4p)g1, (l4p)l3, 0, 0);
}
__device__ __forceinline__ void async_wait() {
#if __has_builtin(__builtin_amdgcn_s_wait_asynccnt)
  __builtin_amdgcn_s_wait_asynccnt(0);
#else
  asm volatile("s_wait_asynccnt 0" ::: "memory");
#endif
}
#endif

__device__ __forceinline__ u16 f2bf(float f) {
  unsigned u = __float_as_uint(f);
  u += 0x7FFFu + ((u >> 16) & 1u);      // round-to-nearest-even
  return (u16)(u >> 16);
}

// 16x32 bf16 fragment (A layout; B^T rows load identically):
// lane l -> row (l&15), K halves (l>>4)*8 and (l>>4)*8+16 : two 16B loads.
__device__ __forceinline__ v16bf load_frag(const u16* base, int stride, int lane) {
  const int row = lane & 15;
  const int kh  = (lane >> 4) << 3;
  const u16* p = base + row * stride + kh;
  union { uint4 q[2]; v16bf v; } u;
  u.q[0] = *(const uint4*)(p);
  u.q[1] = *(const uint4*)(p + 16);
  return u.v;
}

__device__ __forceinline__ v8f wmma_bf16(v16bf a, v16bf b, v8f c) {
  return __builtin_amdgcn_wmma_f32_16x16x32_bf16(
      /*neg_a=*/false, a, /*neg_b=*/false, b,
      /*c_mod=*/(short)0, c, /*reuse_a=*/false, /*reuse_b=*/false);
}

// ---------------------------------------------------------------------------
// fp32 -> bf16 convert (vectorized)
// ---------------------------------------------------------------------------
__global__ __launch_bounds__(256) void convert_bf16_kernel(
    const float* __restrict__ src, u16* __restrict__ dst, int n) {
  int i = (blockIdx.x * 256 + threadIdx.x) * 4;
  if (i + 3 < n) {
    float4 v = *(const float4*)&src[i];
    u16 o[4] = { f2bf(v.x), f2bf(v.y), f2bf(v.z), f2bf(v.w) };
    *(uint2*)&dst[i] = *(const uint2*)o;
  } else {
    for (; i < n; i++) dst[i] = f2bf(src[i]);
  }
}

// src[K][N] fp32 -> dst[N][K] bf16  (weights become B^T so B-fragments load
// with the same contiguous pattern as A-fragments)
__global__ __launch_bounds__(256) void transpose_bf16_kernel(
    const float* __restrict__ src, u16* __restrict__ dst, int K, int N) {
  long long idx = (long long)blockIdx.x * 256 + threadIdx.x;
  if (idx >= (long long)K * N) return;
  int n = (int)(idx / K);
  int k = (int)(idx % K);
  dst[idx] = f2bf(src[(size_t)k * N + n]);
}

// ---------------------------------------------------------------------------
// C[M][N] = A[M][K] * Bt[N][K]^T   (bf16 in, fp32 accum, bf16 or fp32 out)
// 128x128 block tile, BK=32, 8 waves each owning a 32x64 sub-tile.
// Double-buffered LDS; async global->LDS copy overlapped with WMMA.
// ---------------------------------------------------------------------------
template <bool OUT_BF16>
__global__ __launch_bounds__(256) void gemm_bt_kernel(
    const u16* __restrict__ A, const u16* __restrict__ Bt,
    void* __restrict__ Cout, int M, int N, int K) {
  __shared__ u16 sA[2][128 * 32];
  __shared__ u16 sB[2][128 * 32];
  const int tid  = threadIdx.x;
  const int lane = tid & 31;
  const int w    = tid >> 5;
  const int wm   = w >> 1;              // 0..3 : 32-row strip
  const int wn   = w & 1;               // 0..1 : 64-col strip
  const int m0   = blockIdx.y * 128;
  const int n0   = blockIdx.x * 128;
  const int r0   = tid >> 2;            // staging row (and r0+64)
  const int c80  = (tid & 3) << 3;      // staging col (8 bf16 = 16B)

  v8f acc[2][4];
#pragma unroll
  for (int i = 0; i < 2; i++)
#pragma unroll
    for (int j = 0; j < 4; j++) acc[i][j] = (v8f){0,0,0,0,0,0,0,0};

  auto compute = [&](int buf) {
    v16bf af[2], bg[4];
#pragma unroll
    for (int i = 0; i < 2; i++)
      af[i] = load_frag(&sA[buf][(wm * 32 + i * 16) * 32], 32, lane);
#pragma unroll
    for (int j = 0; j < 4; j++)
      bg[j] = load_frag(&sB[buf][(wn * 64 + j * 16) * 32], 32, lane);
#pragma unroll
    for (int j = 0; j < 4; j++)
#pragma unroll
      for (int i = 0; i < 2; i++) acc[i][j] = wmma_bf16(af[i], bg[j], acc[i][j]);
  };

#if defined(ASYNC_COPY)
  auto stage = [&](int buf, int k0) {
    async_cp16(&A [(size_t)(m0 + r0     ) * K + k0 + c80], &sA[buf][(r0     ) * 32 + c80]);
    async_cp16(&A [(size_t)(m0 + r0 + 64) * K + k0 + c80], &sA[buf][(r0 + 64) * 32 + c80]);
    async_cp16(&Bt[(size_t)(n0 + r0     ) * K + k0 + c80], &sB[buf][(r0     ) * 32 + c80]);
    async_cp16(&Bt[(size_t)(n0 + r0 + 64) * K + k0 + c80], &sB[buf][(r0 + 64) * 32 + c80]);
  };
  stage(0, 0);
  async_wait();
  __syncthreads();
  int buf = 0;
  for (int k0 = 0; k0 < K; k0 += 32) {
    const bool more = (k0 + 32 < K);
    if (more) stage(buf ^ 1, k0 + 32);   // streams into other buffer during WMMA
    compute(buf);
    if (more) async_wait();
    __syncthreads();
    buf ^= 1;
  }
#else
  // register-staged double buffer: bulk loads into distinct regs, stores next
  // iteration -> no per-load s_wait_loadcnt 0 serialization.
  uint4 ra0, ra1, rb0, rb1;
  auto gload = [&](int k0) {
    ra0 = *(const uint4*)&A [(size_t)(m0 + r0     ) * K + k0 + c80];
    ra1 = *(const uint4*)&A [(size_t)(m0 + r0 + 64) * K + k0 + c80];
    rb0 = *(const uint4*)&Bt[(size_t)(n0 + r0     ) * K + k0 + c80];
    rb1 = *(const uint4*)&Bt[(size_t)(n0 + r0 + 64) * K + k0 + c80];
  };
  auto sstore = [&](int buf) {
    *(uint4*)&sA[buf][(r0     ) * 32 + c80] = ra0;
    *(uint4*)&sA[buf][(r0 + 64) * 32 + c80] = ra1;
    *(uint4*)&sB[buf][(r0     ) * 32 + c80] = rb0;
    *(uint4*)&sB[buf][(r0 + 64) * 32 + c80] = rb1;
  };
  gload(0);
  int buf = 0;
  for (int k0 = 0; k0 < K; k0 += 32) {
    sstore(buf);
    __syncthreads();
    if (k0 + 32 < K) gload(k0 + 32);     // issues early, overlaps with WMMA
    compute(buf);
    buf ^= 1;
  }
#endif

  const int hi   = lane >> 4;
  const int col0 = lane & 15;
#pragma unroll
  for (int i = 0; i < 2; i++)
#pragma unroll
    for (int j = 0; j < 4; j++) {
      int n = n0 + wn * 64 + j * 16 + col0;
#pragma unroll
      for (int r = 0; r < 8; r++) {
        int m = m0 + wm * 32 + i * 16 + r + 8 * hi;
        float v = acc[i][j][r];
        if constexpr (OUT_BF16) ((u16*)Cout)[(size_t)m * N + n]   = f2bf(v);
        else                    ((float*)Cout)[(size_t)m * N + n] = v;
      }
    }
}

// ---------------------------------------------------------------------------
// Fused sliding-window attention. Block = (128-query tile, head).
// Wave owns a 16-query strip; Q fragments resident in VGPRs; loop over
// 64-key chunks of the 1024-key blocked window with online softmax.
// ---------------------------------------------------------------------------
__global__ __launch_bounds__(256) void attn_kernel(
    const u16* __restrict__ qkv,   // [SEQ][NQK] bf16 (q|k|v per token)
    u16* __restrict__ attn) {      // [SEQ][HQ*HD] bf16
  __shared__ u16 sKV[128 * 64];    // K chunk [64][128] then Vt [128][64]
  __shared__ u16 sP[8][16 * 64];   // per-wave P tile (bf16)

  const int tid   = threadIdx.x;
  const int lane  = tid & 31;
  const int w     = tid >> 5;
  const int h     = blockIdx.y;
  const int qbase = blockIdx.x * 128;
  const int blk   = qbase >> 9;               // 512-token block index
  const int qoff  = h * HD;
  const int koff  = HQ * HD + (h >> 2) * HD;  // GQA: kv head = h/4
  const int voff  = HQ * HD + HKV * HD + (h >> 2) * HD;
  const int hi    = lane >> 4;
  const int col0  = lane & 15;

  // Q fragments for this wave's 16-query strip (4 frags cover hd=128)
  v16bf qf[4];
  {
    const u16* qb = qkv + (size_t)(qbase + w * 16) * NQK + qoff;
#pragma unroll
    for (int kk = 0; kk < 4; kk++) qf[kk] = load_frag(qb + kk * 32, NQK, lane);
  }

  v8f O[8];
#pragma unroll
  for (int f = 0; f < 8; f++) O[f] = (v8f){0,0,0,0,0,0,0,0};
  float mrow[8], lrow[8];
#pragma unroll
  for (int r = 0; r < 8; r++) { mrow[r] = -1e30f; lrow[r] = 0.f; }

  const float scale = 1.0f / (float)HD;       // reference's effective /hd

  for (int jc = 0; jc < 16; jc++) {
    const int kg0 = blk * 512 - 512 + jc * 64;
    if (kg0 < 0) continue;                          // block 0: no prev keys
    if (kg0 > qbase + 127) continue;                // entirely in the future
    if (kg0 + 63 < qbase - (WIN - 1)) continue;     // entirely too old

    // stage K chunk [64 keys][128 hd]
#if defined(ASYNC_COPY)
#pragma unroll
    for (int p = 0; p < 4; p++) {
      int id = p * 256 + tid;
      int r  = id >> 4;
      int c8 = (id & 15) << 3;
      async_cp16(&qkv[(size_t)(kg0 + r) * NQK + koff + c8], &sKV[r * 128 + c8]);
    }
    async_wait();
#else
#pragma unroll
    for (int p = 0; p < 4; p++) {
      int id = p * 256 + tid;
      int r  = id >> 4;
      int c8 = (id & 15) << 3;
      *(uint4*)&sKV[r * 128 + c8] =
          *(const uint4*)&qkv[(size_t)(kg0 + r) * NQK + koff + c8];
    }
#endif
    __syncthreads();

    // S = Q @ K^T : key rows ARE B^T rows -> same fragment loader
    v8f sf[4];
#pragma unroll
    for (int n = 0; n < 4; n++) {
      sf[n] = (v8f){0,0,0,0,0,0,0,0};
#pragma unroll
      for (int kk = 0; kk < 4; kk++) {
        v16bf bk = load_frag(&sKV[(n * 16) * 128 + kk * 32], 128, lane);
        sf[n] = wmma_bf16(qf[kk], bk, sf[n]);
      }
    }

    // band mask + scale in C-fragment layout
    float pv[4][8];
#pragma unroll
    for (int n = 0; n < 4; n++) {
      int kg = kg0 + n * 16 + col0;
#pragma unroll
      for (int r = 0; r < 8; r++) {
        int qg = qbase + w * 16 + r + 8 * hi;
        float s = sf[n][r] * scale;
        bool valid = (kg <= qg) && (kg > qg - WIN);
        pv[n][r] = valid ? s : -1e30f;
      }
    }

    // online softmax per row slot (16-lane shfl_xor reductions)
#pragma unroll
    for (int r = 0; r < 8; r++) {
      float mx = fmaxf(fmaxf(pv[0][r], pv[1][r]), fmaxf(pv[2][r], pv[3][r]));
#pragma unroll
      for (int off = 1; off < 16; off <<= 1)
        mx = fmaxf(mx, __shfl_xor(mx, off, 32));
      float mnew = fmaxf(mrow[r], mx);
      float corr = __expf(mrow[r] - mnew);
      mrow[r] = mnew;
      lrow[r] *= corr;
#pragma unroll
      for (int f = 0; f < 8; f++) O[f][r] *= corr;
      float ls = 0.f;
#pragma unroll
      for (int n = 0; n < 4; n++) {
        float p = (pv[n][r] < -1e29f) ? 0.f : __expf(pv[n][r] - mnew);
        pv[n][r] = p;
        ls += p;
      }
#pragma unroll
      for (int off = 1; off < 16; off <<= 1)
        ls += __shfl_xor(ls, off, 32);
      lrow[r] += ls;
    }

    // P -> LDS (bf16, row-major [16][64]) to reshape into A fragments
#pragma unroll
    for (int n = 0; n < 4; n++)
#pragma unroll
      for (int r = 0; r < 8; r++)
        sP[w][(r + 8 * hi) * 64 + n * 16 + col0] = f2bf(pv[n][r]);
    __syncthreads();   // all waves done reading K chunk from sKV

    // stage V transposed: sKV becomes Vt [128 hd][64 keys]
#pragma unroll
    for (int p = 0; p < 4; p++) {
      int id = p * 256 + tid;
      int r  = id >> 4;          // key 0..63
      int c8 = (id & 15) << 3;   // hd base
      uint4 dv = *(const uint4*)&qkv[(size_t)(kg0 + r) * NQK + voff + c8];
      const u16* e = (const u16*)&dv;
#pragma unroll
      for (int t = 0; t < 8; t++) sKV[(c8 + t) * 64 + r] = e[t];
    }
    __syncthreads();

    // O += P @ V  (A = P frags, B cols = hd -> Vt rows)
    v16bf pa[2];
#pragma unroll
    for (int t = 0; t < 2; t++) pa[t] = load_frag(&sP[w][t * 32], 64, lane);
#pragma unroll
    for (int f = 0; f < 8; f++)
#pragma unroll
      for (int t = 0; t < 2; t++) {
        v16bf bv = load_frag(&sKV[(f * 16) * 64 + t * 32], 64, lane);
        O[f] = wmma_bf16(pa[t], bv, O[f]);
      }
    __syncthreads();   // before next chunk overwrites sKV
  }

  // epilogue: normalize and store bf16 attention output
#pragma unroll
  for (int r = 0; r < 8; r++) {
    float inv = lrow[r] > 0.f ? 1.0f / lrow[r] : 0.f;
    int qg = qbase + w * 16 + r + 8 * hi;
#pragma unroll
    for (int f = 0; f < 8; f++)
      attn[(size_t)qg * (HQ * HD) + h * HD + f * 16 + col0] =
          f2bf(O[f][r] * inv);
  }
}

// ---------------------------------------------------------------------------
extern "C" void kernel_launch(void* const* d_in, const int* in_sizes, int n_in,
                              void* d_out, int out_size, void* d_ws, size_t ws_size,
                              hipStream_t stream) {
  const float* hs = (const float*)d_in[0];
  const float* Wq = (const float*)d_in[1];
  const float* Wk = (const float*)d_in[2];
  const float* Wv = (const float*)d_in[3];
  const float* Wo = (const float*)d_in[4];

  // workspace layout (bf16 / u16 elements), ~139 MB total
  u16* ws    = (u16*)d_ws;
  u16* hsb   = ws;                           // SEQ*DIM
  u16* wqkvt = hsb   + (size_t)SEQ * DIM;    // NQK*DIM   (Wq^T|Wk^T|Wv^T)
  u16* wot   = wqkvt + (size_t)NQK * DIM;    // DIM*DIM   (Wo^T)
  u16* qkvb  = wot   + (size_t)DIM * DIM;    // SEQ*NQK
  u16* attnb = qkvb  + (size_t)SEQ * NQK;    // SEQ*(HQ*HD)

  // 1. hidden_states -> bf16
  {
    int n = SEQ * DIM;
    convert_bf16_kernel<<<n / (256 * 4), 256, 0, stream>>>(hs, hsb, n);
  }
  // 2. weights -> transposed bf16
  {
    long long e;
    e = (long long)DIM * (HQ * HD);
    transpose_bf16_kernel<<<(e + 255) / 256, 256, 0, stream>>>(Wq, wqkvt, DIM, HQ * HD);
    e = (long long)DIM * (HKV * HD);
    transpose_bf16_kernel<<<(e + 255) / 256, 256, 0, stream>>>(
        Wk, wqkvt + (size_t)(HQ * HD) * DIM, DIM, HKV * HD);
    transpose_bf16_kernel<<<(e + 255) / 256, 256, 0, stream>>>(
        Wv, wqkvt + (size_t)(HQ * HD + HKV * HD) * DIM, DIM, HKV * HD);
    e = (long long)(HQ * HD) * DIM;
    transpose_bf16_kernel<<<(e + 255) / 256, 256, 0, stream>>>(Wo, wot, HQ * HD, DIM);
  }
  // 3. fused QKV projection: [SEQ,DIM] x [DIM,NQK] -> bf16
  gemm_bt_kernel<true><<<dim3(NQK / 128, SEQ / 128), 256, 0, stream>>>(
      hsb, wqkvt, qkvb, SEQ, NQK, DIM);
  // 4. sliding-window attention
  attn_kernel<<<dim3(SEQ / 128, HQ), 256, 0, stream>>>(qkvb, attnb);
  // 5. output projection -> fp32
  gemm_bt_kernel<false><<<dim3(DIM / 128, SEQ / 128), 256, 0, stream>>>(
      attnb, wot, d_out, SEQ, DIM, HQ * HD);
}
